// ProbSparseSelfAttention_77275051589845
// MI455X (gfx1250) — matrix operand order repaired and checked by reading, not compiled, not run on hardware
//
#include <hip/hip_runtime.h>
#include <math.h>

#ifndef __has_builtin
#define __has_builtin(x) 0
#endif

#if defined(__gfx1250__) && __has_builtin(__builtin_amdgcn_tensor_load_to_lds) && \
    __has_builtin(__builtin_amdgcn_s_wait_tensorcnt)
#define HAS_TDM 1
#else
#define HAS_TDM 0
#endif

typedef __attribute__((ext_vector_type(16))) __bf16 v16bf;
typedef __attribute__((ext_vector_type(8)))  float  v8f;

#define Bc 2
#define Lc 2048
#define Dc 1024
#define Hc 16
#define DHc 64
#define Uc 1228
#define SCALE 0.125f

// A-fragment K index for element e (contiguous order: k = 16*hi + e, matches B fills)
__device__ __forceinline__ int kmapA(int e, int hi) {
    return 16 * hi + e;
}

#if HAS_TDM
typedef unsigned int v4u __attribute__((ext_vector_type(4)));
typedef int          v8i __attribute__((ext_vector_type(8)));
typedef int          v4i __attribute__((ext_vector_type(4)));
typedef __attribute__((address_space(3))) char lds_char_t;

__device__ __forceinline__ unsigned lds_byte_off(void* p) {
    return (unsigned)(unsigned long long)(lds_char_t*)p;
}

// Tensor Data Mover: 2D bf16 tile (tile_x contiguous elems per row, tile_y rows,
// row_stride elems in global), optional LDS pad for bank-conflict-free strides.
// D# layout per CDNA5 ISA ch.8: group0 = {flags, lds_addr, global_addr, type=2},
// group1 = {data_size=2B, pad fields, tensor dims, tile dims, dim0 stride}.
__device__ __forceinline__ void tdm_load_2d_bf16(void* lds_dst, const void* gsrc,
                                                 unsigned tile_x, unsigned tile_y,
                                                 unsigned row_stride,
                                                 unsigned pad_en, unsigned pad_ivl_enc,
                                                 unsigned pad_amt_enc) {
    unsigned long long ga = (unsigned long long)gsrc;
    v4u g0;
    g0.x = 1u;                                                   // count=1, user mode
    g0.y = lds_byte_off(lds_dst);                                // lds_addr (bytes)
    g0.z = (unsigned)ga;                                         // global_addr[31:0]
    g0.w = (unsigned)((ga >> 32) & 0x01FFFFFFu) | 0x80000000u;   // addr[56:32] | type=2
    v8i g1;
    g1[0] = (int)((1u << 16) | (pad_en << 20) | (pad_ivl_enc << 22) | (pad_amt_enc << 25));
    g1[1] = (int)((tile_x & 0xFFFFu) << 16);                     // tensor_dim0[15:0]
    g1[2] = (int)((tile_x >> 16) | ((tile_y & 0xFFFFu) << 16));  // dim0 hi | dim1 lo
    g1[3] = (int)((tile_y >> 16) | (tile_x << 16));              // dim1 hi | tile_dim0
    g1[4] = (int)(tile_y & 0xFFFFu);                             // tile_dim1 (tile_dim2=0)
    g1[5] = (int)row_stride;                                     // dim0_stride[31:0]
    g1[6] = 0;
    g1[7] = 0;
    v4i z4 = {0, 0, 0, 0};
#if __clang_major__ >= 23
    v8i z8 = {0, 0, 0, 0, 0, 0, 0, 0};
    __builtin_amdgcn_tensor_load_to_lds(g0, g1, z4, z4, z8, 0);
#else
    __builtin_amdgcn_tensor_load_to_lds(g0, g1, z4, z4, 0);
#endif
}
#endif  // HAS_TDM

// ---------------------------------------------------------------- cast f32->bf16
__global__ void cast_bf16_kernel(const float* __restrict__ in, __bf16* __restrict__ out, int n) {
    int i = blockIdx.x * 256 + threadIdx.x;
    if (i < n) out[i] = (__bf16)in[i];
}

// ---------------------------------------------------------------- GEMM (bf16 in, f32 accum)
// C[M,N] = A[M,K] @ B[K,N] + bias.  AHEAD: A stored as [B,H,L,DH] head layout.
// TDM path: wave 0 streams double-buffered A/B tiles via tensor_load_to_lds.
template <bool AHEAD, bool OUTF32>
__global__ void gemm_bf16_kernel(const __bf16* __restrict__ A, const __bf16* __restrict__ Bw,
                                 const float* __restrict__ bias,
                                 float* __restrict__ Cf, __bf16* __restrict__ Cb,
                                 int M, int N, int K) {
#if HAS_TDM
    __shared__ __bf16 As[2][64 * 48];   // 64 rows x 32 k, stride 48 (96B, TDM-padded)
    __shared__ __bf16 Bs[2][32 * 72];   // 32 k x 64 n, stride 72 (144B, TDM-padded)
#else
    __shared__ __bf16 As[1][64 * 48];
    __shared__ __bf16 Bs[1][32 * 72];
#endif

    const int tid  = threadIdx.x;
    const int lane = tid & 31;
    const int w    = tid >> 5;          // 8 waves
    const int wm   = w >> 2;            // 0..1
    const int wn   = w & 3;             // 0..3
    const int hi   = lane >> 4;
    const int ln   = lane & 15;
    const int bm   = blockIdx.y * 64;
    const int bn   = blockIdx.x * 64;

    v8f acc0 = {}, acc1 = {};

#if HAS_TDM
    auto stage = [&](int kc, int buf) {
        const __bf16* abase;
        unsigned astride;
        if constexpr (AHEAD) {
            abase = A + ((((size_t)(bm >> 11) * Hc + (kc >> 6)) * Lc + (bm & (Lc - 1))) * DHc +
                         (kc & 63));
            astride = DHc;
        } else {
            abase = A + (size_t)bm * K + kc;
            astride = (unsigned)K;
        }
        // A: 32 elems x 64 rows; pad 8 DW after every 16 DW -> 48-elem LDS stride
        tdm_load_2d_bf16(&As[buf][0], abase, 32, 64, astride, 1, 3, 7);
        // B: 64 elems x 32 rows; pad 4 DW after every 32 DW -> 72-elem LDS stride
        tdm_load_2d_bf16(&Bs[buf][0], Bw + (size_t)kc * N + bn, 64, 32, (unsigned)N, 1, 4, 3);
    };
    if (tid < 32) stage(0, 0);
#endif

    const int nk = K / 32;
    for (int i = 0; i < nk; ++i) {
#if HAS_TDM
        const int cur = i & 1;
        if (tid < 32) {
            if (i + 1 < nk) {
                stage((i + 1) * 32, cur ^ 1);
                __builtin_amdgcn_s_wait_tensorcnt(2);  // current pair landed
            } else {
                __builtin_amdgcn_s_wait_tensorcnt(0);
            }
        }
        __syncthreads();
#else
        const int cur = 0;
        const int kc  = i * 32;
        {
            int row = tid >> 2, cb = (tid & 3) * 8;
            int m = bm + row, k = kc + cb;
            const uint4* src;
            if constexpr (AHEAD) {
                int b = m >> 11, l = m & (Lc - 1), h = k >> 6, dh = k & 63;
                src = (const uint4*)(A + ((((size_t)b * Hc + h) * Lc + l) * DHc + dh));
            } else {
                src = (const uint4*)(A + (size_t)m * K + k);
            }
            *(uint4*)(&As[0][row * 48 + cb]) = *src;
        }
        {
            int row = tid >> 3, cb = (tid & 7) * 8;
            const uint4* src = (const uint4*)(Bw + (size_t)(kc + row) * N + (bn + cb));
            *(uint4*)(&Bs[0][row * 72 + cb]) = *src;
        }
        __syncthreads();
#endif

        v16bf bfrag, a0, a1;
        for (int e = 0; e < 16; ++e)
            bfrag[e] = Bs[cur][(16 * hi + e) * 72 + 16 * wn + ln];
        for (int e = 0; e < 16; ++e) {
            int kk = kmapA(e, hi);
            a0[e] = As[cur][(16 * wm       + ln) * 48 + kk];
            a1[e] = As[cur][(16 * (wm + 2) + ln) * 48 + kk];
        }
        acc0 = __builtin_amdgcn_wmma_f32_16x16x32_bf16(false, a0, false, bfrag, (short)0, acc0, false, false);
        acc1 = __builtin_amdgcn_wmma_f32_16x16x32_bf16(false, a1, false, bfrag, (short)0, acc1, false, false);
        __syncthreads();
    }

    const int n = bn + 16 * wn + ln;
    const float bv = bias ? bias[n] : 0.0f;
    for (int r = 0; r < 8; ++r) {
        int m0 = bm + 16 * wm       + 8 * hi + r;
        int m1 = bm + 16 * (wm + 2) + 8 * hi + r;
        float v0 = acc0[r] + bv, v1 = acc1[r] + bv;
        if constexpr (OUTF32) {
            Cf[(size_t)m0 * N + n] = v0;
            Cf[(size_t)m1 * N + n] = v1;
        } else {
            Cb[(size_t)m0 * N + n] = (__bf16)v0;
            Cb[(size_t)m1 * N + n] = (__bf16)v1;
        }
    }
}

// ---------------------------------------------------------------- sparsity = max - mean of scores
// one wave per (b,h,16-query tile); streams K in 32-key chunks.
// Cross-lane reduction hoisted out of the K loop (per-lane column partials).
__global__ void sparsity_kernel(const __bf16* __restrict__ qb, const __bf16* __restrict__ kb,
                                float* __restrict__ sp) {
#if HAS_TDM
    __shared__ __bf16 ks[2][32 * 64];
#else
    __shared__ __bf16 ks[1][32 * 64];
#endif

    const int blk  = blockIdx.x;
    const int bh   = blk >> 7;            // 32 (b,h) pairs, L/16 = 128 tiles
    const int tile = blk & 127;
    const int b    = bh >> 4;
    const int h    = bh & 15;
    const int l0   = tile * 16;
    const int lane = threadIdx.x;
    const int hi   = lane >> 4;
    const int ln   = lane & 15;

    v16bf a0, a1;
    {
        const __bf16* qrow = qb + ((size_t)(b * Lc + l0 + ln) * Dc + h * DHc);
        for (int e = 0; e < 16; ++e) {
            int kk = kmapA(e, hi);
            a0[e] = qrow[kk];
            a1[e] = qrow[32 + kk];
        }
    }

    float rmax[8], rsum[8];
    for (int r = 0; r < 8; ++r) { rmax[r] = -INFINITY; rsum[r] = 0.0f; }

#if HAS_TDM
    auto stageK = [&](int c0, int buf) {
        tdm_load_2d_bf16(&ks[buf][0], kb + ((size_t)(b * Lc + c0) * Dc + h * DHc),
                         64, 32, Dc, 0, 0, 0);
    };
    stageK(0, 0);
#endif

    for (int c0 = 0; c0 < Lc; c0 += 32) {
#if HAS_TDM
        const int cur = (c0 >> 5) & 1;
        if (c0 + 32 < Lc) {
            stageK(c0 + 32, cur ^ 1);
            __builtin_amdgcn_s_wait_tensorcnt(1);
        } else {
            __builtin_amdgcn_s_wait_tensorcnt(0);
        }
#else
        const int cur = 0;
        {
            const uint4* src = (const uint4*)(kb + ((size_t)(b * Lc + c0 + lane) * Dc + h * DHc));
            uint4* dst = (uint4*)(&ks[0][lane * 64]);
            for (int j = 0; j < 4; ++j) dst[j] = src[j];
        }
        __syncthreads();
#endif

        for (int nt = 0; nt < 2; ++nt) {
            v16bf b0, b1;
            for (int e = 0; e < 16; ++e) {
                b0[e] = ks[cur][(16 * nt + ln) * 64 +      16 * hi + e];
                b1[e] = ks[cur][(16 * nt + ln) * 64 + 32 + 16 * hi + e];
            }
            v8f s = {};
            s = __builtin_amdgcn_wmma_f32_16x16x32_bf16(false, a0, false, b0, (short)0, s, false, false);
            s = __builtin_amdgcn_wmma_f32_16x16x32_bf16(false, a1, false, b1, (short)0, s, false, false);
            for (int r = 0; r < 8; ++r) {
                rmax[r] = fmaxf(rmax[r], s[r]);
                rsum[r] += s[r];
            }
        }
#if !HAS_TDM
        __syncthreads();
#endif
    }

    // single cross-lane reduction (16-lane half-groups hold 16 column partials)
    for (int r = 0; r < 8; ++r) {
        for (int m = 1; m < 16; m <<= 1) {
            rmax[r] = fmaxf(rmax[r], __shfl_xor(rmax[r], m, 32));
            rsum[r] += __shfl_xor(rsum[r], m, 32);
        }
    }
    if (ln == 0) {
        for (int r = 0; r < 8; ++r) {
            int row = l0 + 8 * hi + r;
            sp[(size_t)bh * Lc + row] = SCALE * rmax[r] - SCALE * rsum[r] / (float)Lc;
        }
    }
}

// ---------------------------------------------------------------- top-U via bitonic sort (desc, tie->lower idx)
__device__ __forceinline__ bool before_desc(float va, int ia, float vb, int ib) {
    return (va > vb) || (va == vb && ia < ib);
}

__global__ void topk_kernel(const float* __restrict__ sp, int* __restrict__ topidx) {
    __shared__ float vals[2048];
    __shared__ int   idxs[2048];
    const int bh  = blockIdx.x;
    const int tid = threadIdx.x;

    for (int i = tid; i < 2048; i += 1024) {
        vals[i] = sp[(size_t)bh * Lc + i];
        idxs[i] = i;
    }
    __syncthreads();

    for (int k = 2; k <= 2048; k <<= 1) {
        for (int j = k >> 1; j > 0; j >>= 1) {
            for (int t = tid; t < 2048; t += 1024) {
                int p = t ^ j;
                if (p > t) {
                    bool desc = ((t & k) == 0);
                    bool tFirst = before_desc(vals[t], idxs[t], vals[p], idxs[p]);
                    bool doSwap = desc ? !tFirst : tFirst;
                    if (doSwap) {
                        float fv = vals[t]; vals[t] = vals[p]; vals[p] = fv;
                        int   iv = idxs[t]; idxs[t] = idxs[p]; idxs[p] = iv;
                    }
                }
            }
            __syncthreads();
        }
    }
    for (int i = tid; i < Uc; i += 1024)
        topidx[(size_t)bh * Uc + i] = idxs[i];
}

// ---------------------------------------------------------------- flash attention over selected rows
// one wave per (b,h, 16-selected-query tile); online softmax; double-buffered TDM K/V stream
__global__ void attn_kernel(const __bf16* __restrict__ qb, const __bf16* __restrict__ kb,
                            const __bf16* __restrict__ vb, const int* __restrict__ topidx,
                            __bf16* __restrict__ outh) {
    __shared__ __bf16 qs[16 * 64];
#if HAS_TDM
    __shared__ __bf16 ks[2][32 * 64];
    __shared__ __bf16 vs[2][32 * 64];
#else
    __shared__ __bf16 ks[1][32 * 64];
    __shared__ __bf16 vs[1][32 * 64];
#endif
    __shared__ float  sS[16 * 33];
    __shared__ __bf16 Ps[16 * 32];
    __shared__ float  fac[16], linv[16];
    __shared__ int    ridx[16], vfl[16];

    const int bh   = blockIdx.x;
    const int b    = bh >> 4;
    const int h    = bh & 15;
    const int u0   = blockIdx.y * 16;
    const int lane = threadIdx.x;
    const int hi   = lane >> 4;
    const int ln   = lane & 15;

    if (lane < 16) {
        int u  = u0 + lane;
        int uc = (u < Uc) ? u : (Uc - 1);
        ridx[lane] = topidx[(size_t)bh * Uc + uc];
        vfl[lane]  = (u < Uc);
    }
    __syncthreads();

#if HAS_TDM
    auto stageKV = [&](int c0, int buf) {
        const __bf16* base = kb + ((size_t)(b * Lc + c0) * Dc + h * DHc);
        tdm_load_2d_bf16(&ks[buf][0], base, 64, 32, Dc, 0, 0, 0);
        const __bf16* vbase = vb + ((size_t)(b * Lc + c0) * Dc + h * DHc);
        tdm_load_2d_bf16(&vs[buf][0], vbase, 64, 32, Dc, 0, 0, 0);
    };
    stageKV(0, 0);
#endif

    // gather selected q rows into LDS
    if (lane < 16) {
        const uint4* src = (const uint4*)(qb + ((size_t)(b * Lc + ridx[lane]) * Dc + h * DHc));
        uint4* dst = (uint4*)(&qs[lane * 64]);
        for (int j = 0; j < 4; ++j) dst[j] = src[j];
    }
    __syncthreads();

    v16bf a0, a1;
    for (int e = 0; e < 16; ++e) {
        int kk = kmapA(e, hi);
        a0[e] = qs[ln * 64 + kk];
        a1[e] = qs[ln * 64 + 32 + kk];
    }

    v8f o0 = {}, o1 = {}, o2 = {}, o3 = {};
    float mrow = -INFINITY, lrow = 0.0f;   // owned by lanes 0..15 (row = lane)

    for (int c0 = 0; c0 < Lc; c0 += 32) {
#if HAS_TDM
        const int cur = (c0 >> 5) & 1;
        if (c0 + 32 < Lc) {
            stageKV(c0 + 32, cur ^ 1);
            __builtin_amdgcn_s_wait_tensorcnt(2);   // current K/V pair landed
        } else {
            __builtin_amdgcn_s_wait_tensorcnt(0);
        }
#else
        const int cur = 0;
        {
            const uint4* srck = (const uint4*)(kb + ((size_t)(b * Lc + c0 + lane) * Dc + h * DHc));
            const uint4* srcv = (const uint4*)(vb + ((size_t)(b * Lc + c0 + lane) * Dc + h * DHc));
            uint4* dk = (uint4*)(&ks[0][lane * 64]);
            uint4* dv = (uint4*)(&vs[0][lane * 64]);
            for (int j = 0; j < 4; ++j) { dk[j] = srck[j]; dv[j] = srcv[j]; }
        }
        __syncthreads();
#endif

        // S = scale * q_sel @ K^T for 32 keys
        for (int nt = 0; nt < 2; ++nt) {
            v16bf b0, b1;
            for (int e = 0; e < 16; ++e) {
                b0[e] = ks[cur][(16 * nt + ln) * 64 +      16 * hi + e];
                b1[e] = ks[cur][(16 * nt + ln) * 64 + 32 + 16 * hi + e];
            }
            v8f s = {};
            s = __builtin_amdgcn_wmma_f32_16x16x32_bf16(false, a0, false, b0, (short)0, s, false, false);
            s = __builtin_amdgcn_wmma_f32_16x16x32_bf16(false, a1, false, b1, (short)0, s, false, false);
            for (int r = 0; r < 8; ++r)
                sS[(8 * hi + r) * 33 + 16 * nt + ln] = SCALE * s[r];
        }
        __syncthreads();

        // online softmax update (row per lane 0..15)
        if (lane < 16) {
            float mnew = mrow;
            for (int j = 0; j < 32; ++j) mnew = fmaxf(mnew, sS[lane * 33 + j]);
            float f = __expf(mrow - mnew);
            float psum = 0.0f;
            for (int j = 0; j < 32; ++j) {
                float p = __expf(sS[lane * 33 + j] - mnew);
                psum += p;
                Ps[lane * 32 + j] = (__bf16)p;
            }
            lrow = lrow * f + psum;
            mrow = mnew;
            fac[lane] = f;
        }
        __syncthreads();

        // rescale accumulators, then O += P @ V
        for (int r = 0; r < 8; ++r) {
            float f = fac[8 * hi + r];
            o0[r] *= f; o1[r] *= f; o2[r] *= f; o3[r] *= f;
        }
        v16bf ap;
        for (int e = 0; e < 16; ++e) ap[e] = Ps[ln * 32 + kmapA(e, hi)];

        v16bf bv0, bv1, bv2, bv3;
        for (int e = 0; e < 16; ++e) {
            int kr = (16 * hi + e) * 64;
            bv0[e] = vs[cur][kr +  0 + ln];
            bv1[e] = vs[cur][kr + 16 + ln];
            bv2[e] = vs[cur][kr + 32 + ln];
            bv3[e] = vs[cur][kr + 48 + ln];
        }
        o0 = __builtin_amdgcn_wmma_f32_16x16x32_bf16(false, ap, false, bv0, (short)0, o0, false, false);
        o1 = __builtin_amdgcn_wmma_f32_16x16x32_bf16(false, ap, false, bv1, (short)0, o1, false, false);
        o2 = __builtin_amdgcn_wmma_f32_16x16x32_bf16(false, ap, false, bv2, (short)0, o2, false, false);
        o3 = __builtin_amdgcn_wmma_f32_16x16x32_bf16(false, ap, false, bv3, (short)0, o3, false, false);
        __syncthreads();
    }

    if (lane < 16) linv[lane] = 1.0f / lrow;
    __syncthreads();

    for (int r = 0; r < 8; ++r) {
        int m = 8 * hi + r;
        if (!vfl[m]) continue;
        size_t base = ((size_t)bh * Lc + ridx[m]) * DHc;
        float li = linv[m];
        outh[base +  0 + ln] = (__bf16)(o0[r] * li);
        outh[base + 16 + ln] = (__bf16)(o1[r] * li);
        outh[base + 32 + ln] = (__bf16)(o2[r] * li);
        outh[base + 48 + ln] = (__bf16)(o3[r] * li);
    }
}

// ---------------------------------------------------------------- host side
extern "C" void kernel_launch(void* const* d_in, const int* in_sizes, int n_in,
                              void* d_out, int out_size, void* d_ws, size_t ws_size,
                              hipStream_t stream) {
    const float* x  = (const float*)d_in[0];
    const float* Wq = (const float*)d_in[1];
    const float* bq = (const float*)d_in[2];
    const float* Wk = (const float*)d_in[3];
    const float* bk = (const float*)d_in[4];
    const float* Wv = (const float*)d_in[5];
    const float* bv = (const float*)d_in[6];
    const float* Wo = (const float*)d_in[7];
    const float* bo = (const float*)d_in[8];

    const int M = Bc * Lc;        // 4096
    const int N = Dc;             // 1024
    const int K = Dc;             // 1024
    const size_t nx = (size_t)M * Dc;   // 4,194,304
    const size_t nw = (size_t)Dc * Dc;  // 1,048,576

    size_t off = 0;
    auto alloc = [&](size_t bytes) {
        void* p = (char*)d_ws + off;
        off = (off + bytes + 255) & ~(size_t)255;
        return p;
    };
    __bf16* xb   = (__bf16*)alloc(nx * 2);
    __bf16* wqb  = (__bf16*)alloc(nw * 2);
    __bf16* wkb  = (__bf16*)alloc(nw * 2);
    __bf16* wvb  = (__bf16*)alloc(nw * 2);
    __bf16* wob  = (__bf16*)alloc(nw * 2);
    __bf16* qbuf = (__bf16*)alloc(nx * 2);
    __bf16* kbuf = (__bf16*)alloc(nx * 2);
    __bf16* vbuf = (__bf16*)alloc(nx * 2);
    float*  sp   = (float*)alloc((size_t)Bc * Hc * Lc * 4);
    int*    tidx = (int*)alloc((size_t)Bc * Hc * Uc * 4);
    __bf16* outh = (__bf16*)alloc(nx * 2);

    // 1. casts
    cast_bf16_kernel<<<(nx + 255) / 256, 256, 0, stream>>>(x,  xb,  (int)nx);
    cast_bf16_kernel<<<(nw + 255) / 256, 256, 0, stream>>>(Wq, wqb, (int)nw);
    cast_bf16_kernel<<<(nw + 255) / 256, 256, 0, stream>>>(Wk, wkb, (int)nw);
    cast_bf16_kernel<<<(nw + 255) / 256, 256, 0, stream>>>(Wv, wvb, (int)nw);
    cast_bf16_kernel<<<(nw + 255) / 256, 256, 0, stream>>>(Wo, wob, (int)nw);

    // 2. QKV projections (bf16 out)
    dim3 gg(N / 64, M / 64);
    gemm_bf16_kernel<false, false><<<gg, 256, 0, stream>>>(xb, wqb, bq, nullptr, qbuf, M, N, K);
    gemm_bf16_kernel<false, false><<<gg, 256, 0, stream>>>(xb, wkb, bk, nullptr, kbuf, M, N, K);
    gemm_bf16_kernel<false, false><<<gg, 256, 0, stream>>>(xb, wvb, bv, nullptr, vbuf, M, N, K);

    // 3. sparsity metric (streamed, no [L,L] materialization)
    sparsity_kernel<<<Bc * Hc * (Lc / 16), 32, 0, stream>>>(qbuf, kbuf, sp);

    // 4. top-U selection
    topk_kernel<<<Bc * Hc, 1024, 0, stream>>>(sp, tidx);

    // 5. flash attention on selected rows, scattered into zeroed head buffer
    hipMemsetAsync(outh, 0, nx * 2, stream);
    dim3 ga(Bc * Hc, (Uc + 15) / 16);
    attn_kernel<<<ga, 32, 0, stream>>>(qbuf, kbuf, vbuf, tidx, outh);

    // 6. output projection (fp32 out, head-layout A)
    gemm_bf16_kernel<true, true><<<gg, 256, 0, stream>>>(outh, wob, bo, (float*)d_out, nullptr, M, N, K);
}
